// GraphClassifier_7249904795692
// MI455X (gfx1250) — matrix operand order repaired and checked by reading, not compiled, not run on hardware
//
#include <hip/hip_runtime.h>
#include <hip/hip_bf16.h>

typedef __attribute__((ext_vector_type(16))) __bf16         v16bf;
typedef __attribute__((ext_vector_type(16))) unsigned short v16u;
typedef __attribute__((ext_vector_type(8)))  unsigned short v8u;
typedef __attribute__((ext_vector_type(8)))  float          v8f;

__device__ __forceinline__ unsigned short f32_to_bf16_bits(float f) {
    unsigned int u = __builtin_bit_cast(unsigned int, f);
    u += 0x7FFFu + ((u >> 16) & 1u);   // round-to-nearest-even
    return (unsigned short)(u >> 16);
}

__device__ __forceinline__ v8u cvt8(float4 a, float4 b) {
    v8u r;
    r[0] = f32_to_bf16_bits(a.x); r[1] = f32_to_bf16_bits(a.y);
    r[2] = f32_to_bf16_bits(a.z); r[3] = f32_to_bf16_bits(a.w);
    r[4] = f32_to_bf16_bits(b.x); r[5] = f32_to_bf16_bits(b.y);
    r[6] = f32_to_bf16_bits(b.z); r[7] = f32_to_bf16_bits(b.w);
    return r;
}

// ---------------------------------------------------------------------------
// WMMA GEMM: Out[M,N] = A[M,K] * W[K,N]; if UseBias, A' = relu(A + biasK[k]).
// Block = 256 threads = 8 waves (4x2); block tile 128x64; K-step 32.
// Each wave computes a 2x2 grid of 16x16 WMMA tiles (4 wmma / K-step).
// LDS tiles are stored in WMMA *fragment order*: frag[f][lane][j] so that a
// fragment read is one contiguous 32B load and loader stores are b128.
// Fragment index math (ISA 7.12.2, 16-bit A 16x32): k = {k4,k3,k2,k1,k0} ->
//   lane_half = k3, j = k2:0 | (k4<<3)  => 8 consecutive k == 8 consecutive j.
// ---------------------------------------------------------------------------
template <bool UseBias>
__global__ __launch_bounds__(256)
void gcn_gemm_bf16(const float* __restrict__ A, const float* __restrict__ W,
                   float* __restrict__ Out, const float* __restrict__ biasK,
                   int M, int N, int K) {
    __shared__ __align__(16) unsigned short Afrag[8][32][16];  // 8 KB: 128 rows
    __shared__ __align__(16) unsigned short Bfrag[4][32][16];  // 4 KB: 64 cols

    const int tid  = threadIdx.x;
    const int lane = tid & 31;
    const int wave = tid >> 5;            // 0..7
    const int wm   = wave & 3;            // row-wave  (32 rows each)
    const int wn   = wave >> 2;           // col-wave  (32 cols each)
    const int rowBase = blockIdx.x * 128;
    const int colBase = blockIdx.y * 64;
    const int hi   = lane >> 4;

    // A loader: thread owns row r, 16 consecutive k starting at cHalf
    const int la_r  = tid >> 1;           // 0..127
    const int la_c  = (tid & 1) * 16;     // 0 or 16
    const int la_fm = la_r >> 4;          // fragment 0..7
    const int la_ln = la_r & 15;          // lane (low half)
    const int la_jb = la_c >> 1;          // j base: 0 or 8

    // B loader: thread owns col, 8 consecutive k starting at lb_r
    const int lb_col = tid & 63;          // 0..63
    const int lb_r   = (tid >> 6) * 8;    // 0,8,16,24
    const int lb_fn  = lb_col >> 4;       // fragment 0..3
    const int lb_ln  = (((lb_r >> 3) & 1) << 4) | (lb_col & 15);
    const int lb_jb  = (lb_r >> 1) & 8;   // 0 or 8

    v8f acc00 = {}, acc01 = {}, acc10 = {}, acc11 = {};

    for (int k0 = 0; k0 < K; k0 += 32) {
        // ---- A tile: 128x32, 16 floats/thread, 4x global_load_b128 ----
        {
            const float4* ap = (const float4*)(A + (size_t)(rowBase + la_r) * K + (k0 + la_c));
            float4 f0 = ap[0], f1 = ap[1], f2 = ap[2], f3 = ap[3];
            if (UseBias) {
                const float4* bp = (const float4*)(biasK + k0 + la_c);
                float4 b0 = bp[0], b1 = bp[1], b2 = bp[2], b3 = bp[3];
                f0.x = fmaxf(f0.x + b0.x, 0.f); f0.y = fmaxf(f0.y + b0.y, 0.f);
                f0.z = fmaxf(f0.z + b0.z, 0.f); f0.w = fmaxf(f0.w + b0.w, 0.f);
                f1.x = fmaxf(f1.x + b1.x, 0.f); f1.y = fmaxf(f1.y + b1.y, 0.f);
                f1.z = fmaxf(f1.z + b1.z, 0.f); f1.w = fmaxf(f1.w + b1.w, 0.f);
                f2.x = fmaxf(f2.x + b2.x, 0.f); f2.y = fmaxf(f2.y + b2.y, 0.f);
                f2.z = fmaxf(f2.z + b2.z, 0.f); f2.w = fmaxf(f2.w + b2.w, 0.f);
                f3.x = fmaxf(f3.x + b3.x, 0.f); f3.y = fmaxf(f3.y + b3.y, 0.f);
                f3.z = fmaxf(f3.z + b3.z, 0.f); f3.w = fmaxf(f3.w + b3.w, 0.f);
            }
            // k = la_c..la_c+7 -> lane-half 0 ; k = la_c+8..la_c+15 -> lane-half 1
            *(v8u*)&Afrag[la_fm][la_ln     ][la_jb] = cvt8(f0, f1);  // ds_store_b128
            *(v8u*)&Afrag[la_fm][la_ln + 16][la_jb] = cvt8(f2, f3);  // ds_store_b128
        }
        // ---- B tile: 32x64, one column strip of 8 k per thread ----
        {
            const float* wp = W + (size_t)(k0 + lb_r) * N + (colBase + lb_col);
            v8u pk;
            #pragma unroll
            for (int i = 0; i < 8; ++i)
                pk[i] = f32_to_bf16_bits(wp[(size_t)i * N]);   // coalesced across lanes
            *(v8u*)&Bfrag[lb_fn][lb_ln][lb_jb] = pk;           // ds_store_b128
        }
        __syncthreads();

        // ---- fragment reads: contiguous 32B each (2x ds_load_b128) ----
        v16bf a0 = __builtin_bit_cast(v16bf, *(const v16u*)&Afrag[wm * 2 + 0][lane][0]);
        v16bf a1 = __builtin_bit_cast(v16bf, *(const v16u*)&Afrag[wm * 2 + 1][lane][0]);
        v16bf b0 = __builtin_bit_cast(v16bf, *(const v16u*)&Bfrag[wn * 2 + 0][lane][0]);
        v16bf b1 = __builtin_bit_cast(v16bf, *(const v16u*)&Bfrag[wn * 2 + 1][lane][0]);

        acc00 = __builtin_amdgcn_wmma_f32_16x16x32_bf16(false, a0, false, b0, (short)0, acc00, false, false);
        acc01 = __builtin_amdgcn_wmma_f32_16x16x32_bf16(false, a0, false, b1, (short)0, acc01, false, false);
        acc10 = __builtin_amdgcn_wmma_f32_16x16x32_bf16(false, a1, false, b0, (short)0, acc10, false, false);
        acc11 = __builtin_amdgcn_wmma_f32_16x16x32_bf16(false, a1, false, b1, (short)0, acc11, false, false);
        __syncthreads();
    }

    // ---- store C: element r -> row (r + hi*8), col lane&15 ----
    const int ocol0 = colBase + wn * 32 + (lane & 15);
    const int orow0 = rowBase + wm * 32 + hi * 8;
    #pragma unroll
    for (int r = 0; r < 8; ++r) {
        float* o0 = Out + (size_t)(orow0 + r) * N;
        float* o1 = Out + (size_t)(orow0 + 16 + r) * N;
        o0[ocol0]      = acc00[r];
        o0[ocol0 + 16] = acc01[r];
        o1[ocol0]      = acc10[r];
        o1[ocol0 + 16] = acc11[r];
    }
}

// ---------------------------------------------------------------------------
// Degree / normalization kernels
// ---------------------------------------------------------------------------
__global__ void deg_init_k(float* deg, int n) {
    int i = blockIdx.x * blockDim.x + threadIdx.x;
    if (i < n) deg[i] = 1.0f;                       // self-loop
}
__global__ void deg_count_k(const long long* __restrict__ dst, float* deg, int E) {
    int e = blockIdx.x * blockDim.x + threadIdx.x;
    if (e < E) unsafeAtomicAdd(&deg[(int)dst[e]], 1.0f);
}
__global__ void dinv_k(const float* __restrict__ deg, float* dinv, int n) {
    int i = blockIdx.x * blockDim.x + threadIdx.x;
    if (i < n) dinv[i] = (deg[i] > 0.0f) ? rsqrtf(deg[i]) : 0.0f;
}

// out[i,f] = dinv[i]^2 * h[i,f]   (self-loop contribution; initializes out)
__global__ void self_init_k(const float* __restrict__ h, const float* __restrict__ dinv,
                            float* __restrict__ out, int total, int F) {
    int idx = blockIdx.x * blockDim.x + threadIdx.x;
    if (idx < total) {
        int i = idx / F;
        float di = dinv[i];
        out[idx] = di * di * h[idx];
    }
}

// out[dst,f] += dinv[src]*dinv[dst] * h[src,f] — F/4 lanes per edge, float4 gather
template <int F>
__global__ __launch_bounds__(256)
void edge_scatter_k(const long long* __restrict__ src, const long long* __restrict__ dst,
                    const float* __restrict__ dinv, const float* __restrict__ h,
                    float* __restrict__ out, int E) {
    constexpr int TPE = F / 4;
    const int t = threadIdx.x;
    const int e = blockIdx.x * (256 / TPE) + t / TPE;
    if (e >= E) return;
    const int s = (int)src[e];
    const int d = (int)dst[e];
    const float norm = dinv[s] * dinv[d];
    const int f = (t % TPE) * 4;
    const float4 hv = *(const float4*)(h + (size_t)s * F + f);
    float* o = out + (size_t)d * F + f;
    unsafeAtomicAdd(o + 0, norm * hv.x);
    unsafeAtomicAdd(o + 1, norm * hv.y);
    unsafeAtomicAdd(o + 2, norm * hv.z);
    unsafeAtomicAdd(o + 3, norm * hv.w);
}

// ---------------------------------------------------------------------------
// Head: per graph g, v = relu(agg2[g*16384+k] + b2[k&127]); logits = v.Wlin+blin;
// log_softmax over 2 classes.
// ---------------------------------------------------------------------------
__global__ __launch_bounds__(256)
void head_k(const float* __restrict__ agg2, const float* __restrict__ b2,
            const float* __restrict__ Wlin, const float* __restrict__ blin,
            float* __restrict__ out) {
    __shared__ float s0[256];
    __shared__ float s1[256];
    const int g = blockIdx.x;
    const int t = threadIdx.x;
    const float* base = agg2 + (size_t)g * 16384;
    float p0 = 0.0f, p1 = 0.0f;
    for (int k = t; k < 16384; k += 256) {
        float v = fmaxf(base[k] + b2[k & 127], 0.0f);
        p0 += v * Wlin[2 * k + 0];
        p1 += v * Wlin[2 * k + 1];
    }
    s0[t] = p0; s1[t] = p1;
    __syncthreads();
    for (int off = 128; off > 0; off >>= 1) {
        if (t < off) { s0[t] += s0[t + off]; s1[t] += s1[t + off]; }
        __syncthreads();
    }
    if (t == 0) {
        float l0 = s0[0] + blin[0];
        float l1 = s1[0] + blin[1];
        float m  = fmaxf(l0, l1);
        float lse = m + logf(expf(l0 - m) + expf(l1 - m));
        out[2 * g + 0] = l0 - lse;
        out[2 * g + 1] = l1 - lse;
    }
}

// ---------------------------------------------------------------------------
extern "C" void kernel_launch(void* const* d_in, const int* in_sizes, int n_in,
                              void* d_out, int out_size, void* d_ws, size_t ws_size,
                              hipStream_t stream) {
    const float*     x    = (const float*)d_in[0];
    const long long* ei   = (const long long*)d_in[1];   // int64 edge_index [2,E]
    const float*     W1   = (const float*)d_in[2];
    const float*     b1   = (const float*)d_in[3];
    const float*     W2   = (const float*)d_in[4];
    const float*     b2   = (const float*)d_in[5];
    const float*     Wlin = (const float*)d_in[6];
    const float*     blin = (const float*)d_in[7];
    float*           out  = (float*)d_out;

    const int K1 = 768, F1 = 256, F2 = 128;
    const int N  = in_sizes[0] / K1;    // 32768 nodes
    const int E  = in_sizes[1] / 2;     // 524288 edges
    const long long* src = ei;
    const long long* dst = ei + E;

    // workspace layout (all fully written before read)
    char*  p    = (char*)d_ws;
    float* deg  = (float*)p; p += (size_t)N * sizeof(float);
    float* dinv = (float*)p; p += (size_t)N * sizeof(float);
    float* h1   = (float*)p; p += (size_t)N * F1 * sizeof(float);
    float* a1   = (float*)p; p += (size_t)N * F1 * sizeof(float);
    float* h2   = (float*)p; p += (size_t)N * F2 * sizeof(float);
    float* a2   = (float*)p; p += (size_t)N * F2 * sizeof(float);
    (void)ws_size; (void)n_in; (void)out_size;

    // degrees + symmetric norm
    deg_init_k <<<(N + 255) / 256, 256, 0, stream>>>(deg, N);
    deg_count_k<<<(E + 255) / 256, 256, 0, stream>>>(dst, deg, E);
    dinv_k     <<<(N + 255) / 256, 256, 0, stream>>>(deg, dinv, N);

    // layer 1: h1 = x @ W1 ; a1 = D^-1/2 A D^-1/2 h1  (bias/relu fused into GEMM2 load)
    dim3 g1(N / 128, F1 / 64);
    gcn_gemm_bf16<false><<<g1, 256, 0, stream>>>(x, W1, h1, nullptr, N, F1, K1);
    self_init_k<<<((size_t)N * F1 + 255) / 256, 256, 0, stream>>>(h1, dinv, a1, N * F1, F1);
    edge_scatter_k<256><<<(E + 3) / 4, 256, 0, stream>>>(src, dst, dinv, h1, a1, E);

    // layer 2: h2 = relu(a1 + b1) @ W2 ; a2 = aggregation(h2)
    dim3 g2(N / 128, F2 / 64);
    gcn_gemm_bf16<true><<<g2, 256, 0, stream>>>(a1, W2, h2, b1, N, F2, F1);
    self_init_k<<<((size_t)N * F2 + 255) / 256, 256, 0, stream>>>(h2, dinv, a2, N * F2, F2);
    edge_scatter_k<128><<<(E + 7) / 8, 256, 0, stream>>>(src, dst, dinv, h2, a2, E);

    // head: 256 graphs of 128 nodes
    head_k<<<N / 128, 256, 0, stream>>>(a2, b2, Wlin, blin, out);
}
